// MaxSimilarityMatcher_61014305407065
// MI455X (gfx1250) — matrix operand order repaired and checked by reading, not compiled, not run on hardware
//
#include <hip/hip_runtime.h>
#include <hip/hip_bf16.h>
#include <math.h>

// Problem constants (fixed by reference setup_inputs)
#define NA 4096
#define NB 4096
#define DD 256
#define NV 9          // 1 identity + 8 steered variants
#define KPROT 8
#define INV_TEMP 20.0f
#define THRESHOLD 0.01f
#define EPS_D 1e-12f
#define NEG_INF (-3.4e38f)

typedef __attribute__((ext_vector_type(16))) __bf16 v16bf;
typedef __attribute__((ext_vector_type(8)))  __bf16 v8bf;
typedef __attribute__((ext_vector_type(8)))  float  v8f;

// ---------------------------------------------------------------------------
// WMMA fragment loaders (layouts per CDNA5 ISA §7.12.2, wave32)
// A fragment: 16x32 bf16, lane&15 = row M, lane>=16 -> K offset 8;
//   v16 elements 0..7 -> K = koff+0..7, elements 8..15 -> K = 16+koff+0..7
// B fragment: 32x16 bf16 (we feed B^T tiles: column n = descriptor row),
//   lane&15 = col N, same K layout as A (lane>=16 -> K offset 8)
// ---------------------------------------------------------------------------
__device__ __forceinline__ v16bf load_frag_a(const __bf16* __restrict__ base, int lane) {
  const int row  = lane & 15;
  const int koff = (lane >> 4) << 3;   // 0 or 8
  const __bf16* p = base + row * DD + koff;
  v8bf lo = *(const v8bf*)(p);         // K = koff .. koff+7
  v8bf hi = *(const v8bf*)(p + 16);    // K = 16+koff .. 16+koff+7
  v16bf r;
#pragma unroll
  for (int i = 0; i < 8; ++i) { r[i] = lo[i]; r[i + 8] = hi[i]; }
  return r;
}

__device__ __forceinline__ v16bf load_frag_b(const __bf16* __restrict__ base, int lane) {
  const int col  = lane & 15;
  const int koff = (lane >> 4) << 3;   // 0 or 8
  const __bf16* p = base + col * DD + koff;
  v8bf lo = *(const v8bf*)(p);
  v8bf hi = *(const v8bf*)(p + 16);
  v16bf r;
#pragma unroll
  for (int i = 0; i < 8; ++i) { r[i] = lo[i]; r[i + 8] = hi[i]; }
  return r;
}

__device__ __forceinline__ v8f wmma_bf16(v16bf a, v16bf b, v8f c) {
  // (neg_a, A, neg_b, B, c_mod, C, reuse_a, reuse_b)
  return __builtin_amdgcn_wmma_f32_16x16x32_bf16(false, a, false, b, (short)0, c,
                                                 false, false);
}

// ---------------------------------------------------------------------------
// K0: f32 -> bf16 row convert + optional row sum-of-squares (256 cols/row)
// ---------------------------------------------------------------------------
__global__ __launch_bounds__(256) void convert_rows_kernel(
    const float* __restrict__ src, __bf16* __restrict__ dst,
    float* __restrict__ sumsq) {
  const int row = blockIdx.x;
  const int t   = threadIdx.x;
  const float x = src[(size_t)row * DD + t];
  const __bf16 h = (__bf16)x;
  dst[(size_t)row * DD + t] = h;
  __shared__ float s[256];
  const float xf = (float)h;
  s[t] = xf * xf;
  __syncthreads();
  for (int off = 128; off; off >>= 1) {
    if (t < off) s[t] += s[t + off];
    __syncthreads();
  }
  if (t == 0 && sumsq) sumsq[row] = s[0];
}

// row sum-of-squares for bf16 rows (steered descriptor norms)
__global__ __launch_bounds__(256) void rowsumsq_bf16_kernel(
    const __bf16* __restrict__ src, float* __restrict__ out) {
  const int row = blockIdx.x;
  const int t   = threadIdx.x;
  const float x = (float)src[(size_t)row * DD + t];
  __shared__ float s[256];
  s[t] = x * x;
  __syncthreads();
  for (int off = 128; off; off >>= 1) {
    if (t < off) s[t] += s[t + off];
    __syncthreads();
  }
  if (t == 0) out[row] = s[0];
}

// ---------------------------------------------------------------------------
// K1: steering GEMM  Aout[k] = A0 @ M_k^T  (4096x256 @ 256x256), bf16 WMMA.
// Block = 8 waves -> 128(n) x 128(e) tile; each wave 32x64 (2x4 WMMA tiles).
// grid = (e_tiles=2, n_tiles=32, k=8)
// ---------------------------------------------------------------------------
__global__ __launch_bounds__(256) void steer_wmma_kernel(
    const __bf16* __restrict__ A0, const __bf16* __restrict__ Mall,
    __bf16* __restrict__ Aout) {
  const int lane = threadIdx.x & 31;
  const int w    = threadIdx.x >> 5;
  const int wr   = w & 3;
  const int wc   = w >> 2;
  const int M0   = blockIdx.y * 128 + wr * 32;
  const int N0   = blockIdx.x * 128 + wc * 64;
  const int mrow = (lane >> 4) << 3;
  const int ncol = lane & 15;
  const __bf16* Mb  = Mall + (size_t)blockIdx.z * DD * DD;
  __bf16*       Out = Aout + (size_t)blockIdx.z * NA * DD;

  v8f acc[2][4] = {};
  for (int kk = 0; kk < DD; kk += 32) {
    v16bf aF[2];
    aF[0] = load_frag_a(A0 + (size_t)(M0     ) * DD + kk, lane);
    aF[1] = load_frag_a(A0 + (size_t)(M0 + 16) * DD + kk, lane);
    v16bf bF[4];
#pragma unroll
    for (int j = 0; j < 4; ++j)
      bF[j] = load_frag_b(Mb + (size_t)(N0 + 16 * j) * DD + kk, lane);
#pragma unroll
    for (int i = 0; i < 2; ++i)
#pragma unroll
      for (int j = 0; j < 4; ++j)
        acc[i][j] = wmma_bf16(aF[i], bF[j], acc[i][j]);
  }
#pragma unroll
  for (int i = 0; i < 2; ++i)
#pragma unroll
    for (int r = 0; r < 8; ++r) {
      const int n = M0 + i * 16 + mrow + r;
#pragma unroll
      for (int j = 0; j < 4; ++j) {
        const int e = N0 + j * 16 + ncol;
        Out[(size_t)n * DD + e] = (__bf16)(acc[i][j][r]);
      }
    }
}

// ---------------------------------------------------------------------------
// K2: main corr kernel. For each 128x128 output tile, loop 9 variants:
// f32-accum bf16 WMMA over K=256, then corr = max_v(-sqrt(a2+b2-2*dot)).
// Block = 8 waves; wave = 32(M) x 64(N) = 2x4 WMMA tiles. grid = (32, 32).
// ---------------------------------------------------------------------------
__global__ __launch_bounds__(256) void corr_wmma_kernel(
    const __bf16* __restrict__ Aall, const __bf16* __restrict__ Bb,
    const float* __restrict__ a2, const float* __restrict__ b2,
    float* __restrict__ corr) {
  const int lane = threadIdx.x & 31;
  const int w    = threadIdx.x >> 5;
  const int wr   = w & 3;
  const int wc   = w >> 2;
  const int M0   = blockIdx.y * 128 + wr * 32;
  const int N0   = blockIdx.x * 128 + wc * 64;
  const int mrow = (lane >> 4) << 3;
  const int ncol = lane & 15;

  float runmax[2][4][8];
#pragma unroll
  for (int i = 0; i < 2; ++i)
#pragma unroll
    for (int j = 0; j < 4; ++j)
#pragma unroll
      for (int r = 0; r < 8; ++r) runmax[i][j][r] = NEG_INF;

  float b2r[4];
#pragma unroll
  for (int j = 0; j < 4; ++j) b2r[j] = b2[N0 + j * 16 + ncol];

  for (int v = 0; v < NV; ++v) {
    const __bf16* Av = Aall + (size_t)v * NA * DD;
    v8f acc[2][4] = {};
    for (int kk = 0; kk < DD; kk += 32) {
      if (kk + 32 < DD) {  // warm L2 for the next k-step (global_prefetch_b8)
        __builtin_prefetch(Av + (size_t)M0 * DD + kk + 32, 0, 0);
        __builtin_prefetch(Bb + (size_t)N0 * DD + kk + 32, 0, 0);
      }
      v16bf aF[2];
      aF[0] = load_frag_a(Av + (size_t)(M0     ) * DD + kk, lane);
      aF[1] = load_frag_a(Av + (size_t)(M0 + 16) * DD + kk, lane);
      v16bf bF[4];
#pragma unroll
      for (int j = 0; j < 4; ++j)
        bF[j] = load_frag_b(Bb + (size_t)(N0 + 16 * j) * DD + kk, lane);
#pragma unroll
      for (int i = 0; i < 2; ++i)
#pragma unroll
        for (int j = 0; j < 4; ++j)
          acc[i][j] = wmma_bf16(aF[i], bF[j], acc[i][j]);
    }
#pragma unroll
    for (int i = 0; i < 2; ++i) {
#pragma unroll
      for (int r = 0; r < 8; ++r) {
        const float a2v = a2[v * NA + M0 + i * 16 + mrow + r];
#pragma unroll
        for (int j = 0; j < 4; ++j) {
          float d2 = a2v + b2r[j] - 2.0f * acc[i][j][r];
          d2 = d2 > 0.0f ? d2 : 0.0f;
          const float c = -sqrtf(d2 + EPS_D);
          runmax[i][j][r] = fmaxf(runmax[i][j][r], c);
        }
      }
    }
  }

#pragma unroll
  for (int i = 0; i < 2; ++i)
#pragma unroll
    for (int r = 0; r < 8; ++r) {
      const int n = M0 + i * 16 + mrow + r;
#pragma unroll
      for (int j = 0; j < 4; ++j) {
        const int m = N0 + j * 16 + ncol;
        corr[(size_t)n * NB + m] = INV_TEMP * runmax[i][j][r];
      }
    }
}

// ---------------------------------------------------------------------------
// K3: per-row max and sum(exp(c - max))
// ---------------------------------------------------------------------------
__global__ __launch_bounds__(256) void row_stats_kernel(
    const float* __restrict__ C, float* __restrict__ rmax, float* __restrict__ rsum) {
  const int n = blockIdx.x;
  const int t = threadIdx.x;
  const float* row = C + (size_t)n * NB;
  __shared__ float s[256];
  float mx = NEG_INF;
  for (int m = t; m < NB; m += 256) mx = fmaxf(mx, row[m]);
  s[t] = mx; __syncthreads();
  for (int off = 128; off; off >>= 1) { if (t < off) s[t] = fmaxf(s[t], s[t + off]); __syncthreads(); }
  const float M = s[0];
  __syncthreads();
  float sm = 0.0f;
  for (int m = t; m < NB; m += 256) sm += expf(row[m] - M);
  s[t] = sm; __syncthreads();
  for (int off = 128; off; off >>= 1) { if (t < off) s[t] += s[t + off]; __syncthreads(); }
  if (t == 0) { rmax[n] = M; rsum[n] = s[0]; }
}

// ---------------------------------------------------------------------------
// K4: per-column max (and optionally sum(exp)). 32 cols/block, 8-way row split.
// ---------------------------------------------------------------------------
__global__ __launch_bounds__(256) void col_stats_kernel(
    const float* __restrict__ C, float* __restrict__ cmax,
    float* __restrict__ csum, int do_sum) {
  const int tx  = threadIdx.x & 31;
  const int ty  = threadIdx.x >> 5;
  const int col = blockIdx.x * 32 + tx;
  __shared__ float s[256];
  float mx = NEG_INF;
  for (int r = ty; r < NA; r += 8) mx = fmaxf(mx, C[(size_t)r * NB + col]);
  s[threadIdx.x] = mx; __syncthreads();
  if (ty == 0) {
#pragma unroll
    for (int q = 1; q < 8; ++q) mx = fmaxf(mx, s[q * 32 + tx]);
    s[tx] = mx;
  }
  __syncthreads();
  const float M = s[tx];
  __syncthreads();
  if (do_sum) {
    float sm = 0.0f;
    for (int r = ty; r < NA; r += 8) sm += expf(C[(size_t)r * NB + col] - M);
    s[threadIdx.x] = sm; __syncthreads();
    if (ty == 0) {
#pragma unroll
      for (int q = 1; q < 8; ++q) sm += s[q * 32 + tx];
      cmax[col] = M; csum[col] = sm;
    }
  } else {
    if (ty == 0) cmax[col] = M;
  }
}

// ---------------------------------------------------------------------------
// K5: P = softmax_col * softmax_row (written to d_out), plus per-row Pmax and
// first-index argmax (jnp.argmax semantics).
// ---------------------------------------------------------------------------
__global__ __launch_bounds__(256) void p_row_kernel(
    const float* __restrict__ C, const float* __restrict__ rmax,
    const float* __restrict__ rsum, const float* __restrict__ cmax,
    const float* __restrict__ csum, float* __restrict__ P,
    float* __restrict__ rowPmax, int* __restrict__ jidx) {
  const int n = blockIdx.x;
  const int t = threadIdx.x;
  const float rm = rmax[n];
  const float ri = 1.0f / rsum[n];
  __shared__ float sv[256];
  __shared__ int   si[256];
  float bv = NEG_INF; int bi = 0x7fffffff;
  for (int m = t; m < NB; m += 256) {
    const float c = C[(size_t)n * NB + m];
    const float p = (expf(c - rm) * ri) * (expf(c - cmax[m]) / csum[m]);
    P[(size_t)n * NB + m] = p;
    if (p > bv) { bv = p; bi = m; }
  }
  sv[t] = bv; si[t] = bi; __syncthreads();
  for (int off = 128; off; off >>= 1) {
    if (t < off) {
      const float ov = sv[t + off]; const int oi = si[t + off];
      if (ov > sv[t] || (ov == sv[t] && oi < si[t])) { sv[t] = ov; si[t] = oi; }
    }
    __syncthreads();
  }
  if (t == 0) { rowPmax[n] = sv[0]; jidx[n] = si[0]; }
}

// ---------------------------------------------------------------------------
// K6: elementwise mutual-NN mask -> float {0,1}
// ---------------------------------------------------------------------------
__global__ __launch_bounds__(256) void mask_kernel(
    const float* __restrict__ P, const float* __restrict__ rowPmax,
    const float* __restrict__ colPmax, float* __restrict__ maskOut) {
  const size_t idx = (size_t)blockIdx.x * 256 + threadIdx.x;
  const int n = (int)(idx >> 12);
  const int m = (int)(idx & (NB - 1));
  const float p = P[idx];
  maskOut[idx] = (p == rowPmax[n] && p == colPmax[m] && p > THRESHOLD) ? 1.0f : 0.0f;
}

// ---------------------------------------------------------------------------
// K7: valid flag at argmax + keypoint gather
// ---------------------------------------------------------------------------
__global__ __launch_bounds__(256) void finalize_kernel(
    const float* __restrict__ P, const float* __restrict__ rowPmax,
    const float* __restrict__ colPmax, const int* __restrict__ jidx,
    const float* __restrict__ kpB, float* __restrict__ valid,
    float* __restrict__ matches) {
  const int n = blockIdx.x * 256 + threadIdx.x;
  if (n >= NA) return;
  const int m = jidx[n];
  const float p = P[(size_t)n * NB + m];
  valid[n] = (p == rowPmax[n] && p == colPmax[m] && p > THRESHOLD) ? 1.0f : 0.0f;
  matches[2 * n]     = kpB[2 * m];
  matches[2 * n + 1] = kpB[2 * m + 1];
}

// ---------------------------------------------------------------------------
// Host launcher
// ---------------------------------------------------------------------------
extern "C" void kernel_launch(void* const* d_in, const int* in_sizes, int n_in,
                              void* d_out, int out_size, void* d_ws, size_t ws_size,
                              hipStream_t stream) {
  (void)in_sizes; (void)n_in; (void)out_size; (void)ws_size;
  // Inputs (setup_inputs order)
  const float* kpA    = (const float*)d_in[0];  (void)kpA;
  const float* descA  = (const float*)d_in[1];
  const float* kpB    = (const float*)d_in[2];
  const float* descB  = (const float*)d_in[3];
  const float* protos = (const float*)d_in[4];

  // Workspace layout (all offsets 256B-aligned)
  char* ws = (char*)d_ws;
  __bf16* a_bf   = (__bf16*)(ws + 0);                       // 9*4096*256*2 = 18,874,368
  __bf16* b_bf   = (__bf16*)(ws + 18874368);                // 4096*256*2   =  2,097,152
  __bf16* m_bf   = (__bf16*)(ws + 20971520);                // 8*256*256*2  =  1,048,576
  float*  corr   = (float*) (ws + 22020096);                // 4096*4096*4  = 67,108,864
  float*  a2     = (float*) (ws + 89128960);                // 9*4096*4     =    147,456
  float*  b2     = (float*) (ws + 89276416);                // 4096*4
  float*  rmax   = (float*) (ws + 89292800);
  float*  rsum   = (float*) (ws + 89309184);
  float*  cmax   = (float*) (ws + 89325568);
  float*  csum   = (float*) (ws + 89341952);
  float*  rowPmx = (float*) (ws + 89358336);
  float*  colPmx = (float*) (ws + 89374720);
  int*    jidx   = (int*)   (ws + 89391104);

  // Outputs: P | mask | valid | matches_B (flat, return order)
  float* P_out    = (float*)d_out;
  float* mask_out = P_out + (size_t)NA * NB;
  float* valid_o  = mask_out + (size_t)NA * NB;
  float* match_o  = valid_o + NA;

  // K0: bf16 conversion + norms
  convert_rows_kernel<<<NB, 256, 0, stream>>>(descB, b_bf, b2);
  convert_rows_kernel<<<NA, 256, 0, stream>>>(descA, a_bf, a2);        // variant 0
  convert_rows_kernel<<<KPROT * DD, 256, 0, stream>>>(protos, m_bf, nullptr);

  // K1: steered descriptors via WMMA (variants 1..8)
  steer_wmma_kernel<<<dim3(DD / 128, NA / 128, KPROT), 256, 0, stream>>>(
      a_bf, m_bf, a_bf + (size_t)NA * DD);

  // steered-row norms (consistent with bf16-rounded values used in the GEMM)
  rowsumsq_bf16_kernel<<<KPROT * NA, 256, 0, stream>>>(a_bf + (size_t)NA * DD,
                                                       a2 + NA);

  // K2: 9-variant max-distance correlation (bf16 WMMA, f32 accumulate)
  corr_wmma_kernel<<<dim3(NB / 128, NA / 128), 256, 0, stream>>>(
      a_bf, b_bf, a2, b2, corr);

  // K3/K4: dual-softmax statistics
  row_stats_kernel<<<NA, 256, 0, stream>>>(corr, rmax, rsum);
  col_stats_kernel<<<NB / 32, 256, 0, stream>>>(corr, cmax, csum, 1);

  // K5: P + per-row max/argmax
  p_row_kernel<<<NA, 256, 0, stream>>>(corr, rmax, rsum, cmax, csum,
                                       P_out, rowPmx, jidx);

  // K6a: per-column max of P
  col_stats_kernel<<<NB / 32, 256, 0, stream>>>(P_out, colPmx, csum, 0);

  // K6b: mutual-NN threshold mask
  mask_kernel<<<(int)(((size_t)NA * NB) / 256), 256, 0, stream>>>(
      P_out, rowPmx, colPmx, mask_out);

  // K7: valid + matched keypoints
  finalize_kernel<<<(NA + 255) / 256, 256, 0, stream>>>(
      P_out, rowPmx, colPmx, jidx, kpB, valid_o, match_o);
}
